// TaskSpecificFeedforwardLayer_11184094839027
// MI455X (gfx1250) — matrix operand order repaired and checked by reading, not compiled, not run on hardware
//
#include <hip/hip_runtime.h>
#include <hip/hip_bf16.h>

// ---------------------------------------------------------------------------
// Fused task-conditioned FFN + residual + LayerNorm for MI455X (gfx1250).
//   h = relu(x @ W1^T + b1[task]);  y = x + h @ W2^T + b2[task];  LN(y)
// Compute-bound (5.5e11 FLOP vs ~300MB HBM) -> bf16 WMMA, fp32 accumulate.
// Round 2: software-pipelined fragment loads (rotating registers) so weight
// global_load_b128s overlap WMMAs instead of serializing on a reused VGPR set;
// global_prefetch of the next H-chunk.
// ---------------------------------------------------------------------------

#define S_DIM 512
#define B_DIM 64
#define D_DIM 1024
#define H_DIM 4096
#define T_DIM 16
#define LN_EPS 1e-5f

#define TN  32              // tokens per workgroup
#define HC  128             // H processed per main-loop iteration
#define LDX (D_DIM + 8)     // bf16 elements: x-tile row stride (pad = 16B)
#define LDH (HC + 8)        // bf16 elements: h-chunk row stride
#define LDY (D_DIM + 8)     // f32 elements: y-tile row stride

typedef __attribute__((ext_vector_type(16))) __bf16 v16bf;
typedef __attribute__((ext_vector_type(4)))  __bf16 v4bf;
typedef __attribute__((ext_vector_type(8)))  float  v8f;

// Dynamic-LDS byte offsets. ytile (epilogue) aliases xb+hb (main loop).
#define XB_OFF   0
#define HB_OFF   (TN * LDX * 2)            // 66048
#define YT_BYTES (TN * LDY * 4)            // 132096  (covers xb+hb regions)
#define TSK_OFF  YT_BYTES
#define RED_OFF  (TSK_OFF + TN * 4)
#define MUS_OFF  (RED_OFF + 256 * 2 * 4)
#define SMEM_BYTES (MUS_OFF + TN * 2 * 4)  // 134528 bytes

// --- WMMA fragment helpers (layouts per CDNA5 ISA 7.12.2, wave32) ----------

// A: 16x32 bf16 tile, element (m = lane&15, k). Lanes 0-15 hold k = 0..7 and
// 16..23; lanes 16-31 hold k = 8..15 and 24..31 (two contiguous 16B chunks).
__device__ __forceinline__ v16bf load_a_frag(const __bf16* tile, int ld, int lane) {
    const int r  = lane & 15;
    const int kb = (lane >> 4) << 3;   // 0 or 8
    const __bf16* p = tile + r * ld + kb;
    union { v16bf v; uint4 q[2]; } u;
    u.q[0] = *(const uint4*)(p);        // k = kb .. kb+7
    u.q[1] = *(const uint4*)(p + 16);   // k = kb+16 .. kb+23
    return u.v;
}

// B: 32x16 bf16 tile, element (k, n = lane&15). Source is "k-contiguous per
// column": column n lives at colmaj + n*ld + k. Lanes 0-15 hold k = 0..7 and
// 16..23; lanes 16-31 hold k = 8..15 and 24..31 (same as A).
__device__ __forceinline__ v16bf load_b_frag(const __bf16* colmaj, int ld, int lane) {
    const int n  = lane & 15;
    const int kb = (lane >> 4) << 3;   // 0 or 8
    const __bf16* p = colmaj + n * ld + kb;
    union { v16bf v; uint4 q[2]; } u;
    u.q[0] = *(const uint4*)(p);
    u.q[1] = *(const uint4*)(p + 16);
    return u.v;
}

__device__ __forceinline__ v8f wmma_bf16(v16bf a, v16bf b, v8f c) {
    return __builtin_amdgcn_wmma_f32_16x16x32_bf16(
        /*neg_a=*/false, a, /*neg_b=*/false, b,
        /*c_mod=*/(short)0, c, /*reuse_a=*/false, /*reuse_b=*/false);
}

// --- weight fp32 -> bf16 pre-convert (weights are reused by every WG) ------
__global__ void cvt_bf16_kernel(const float* __restrict__ src,
                                __bf16* __restrict__ dst, int n4) {
    int i = blockIdx.x * blockDim.x + threadIdx.x;
    if (i < n4) {
        float4 v = ((const float4*)src)[i];
        v4bf o;
        o.x = (__bf16)v.x; o.y = (__bf16)v.y;
        o.z = (__bf16)v.z; o.w = (__bf16)v.w;
        ((v4bf*)dst)[i] = o;
    }
}

// --- fused FFN kernel ------------------------------------------------------
__global__ __launch_bounds__(256)
void ffn_fused_kernel(const int*   __restrict__ task_id,
                      const float* __restrict__ x,
                      const float* __restrict__ b1,
                      const float* __restrict__ b2,
                      const float* __restrict__ ln_w,
                      const float* __restrict__ ln_b,
                      const __bf16* __restrict__ W1b,   // [H, D] bf16
                      const __bf16* __restrict__ W2b,   // [D, H] bf16
                      float* __restrict__ out) {
    extern __shared__ char smem[];
    __bf16* xb    = (__bf16*)(smem + XB_OFF);  // [TN][LDX] bf16 x tile
    __bf16* hb    = (__bf16*)(smem + HB_OFF);  // [TN][LDH] bf16 h chunk
    float*  ytile = (float*)(smem);            // [TN][LDY] f32 (epilogue)
    int*    tsk   = (int*)(smem + TSK_OFF);
    float*  red   = (float*)(smem + RED_OFF);
    float*  musig = (float*)(smem + MUS_OFF);

    const int tid  = threadIdx.x;
    const int lane = tid & 31;
    const int wave = tid >> 5;                 // 0..7
    const int n0   = blockIdx.x * TN;          // first token (row of [S*B, D])

    // ---- prologue: task ids + x tile (fp32 -> bf16 into LDS) ----
    if (tid < TN) {
        int n = n0 + tid;
        tsk[tid] = task_id[n % B_DIM];
    }
    for (int i = tid; i < TN * (D_DIM / 4); i += 256) {
        int row = i >> 8;                      // D/4 == 256
        int c4  = i & 255;
        float4 v = ((const float4*)(x + (size_t)(n0 + row) * D_DIM))[c4];
        v4bf o;
        o.x = (__bf16)v.x; o.y = (__bf16)v.y;
        o.z = (__bf16)v.z; o.w = (__bf16)v.w;
        *(v4bf*)(xb + row * LDX + c4 * 4) = o;
    }
    __syncthreads();

    // task id per C-layout row this lane touches (row = rt*16 + rbase + j)
    const int rbase = (lane >> 4) << 3;
    const int col   = lane & 15;
    int trow[2][8];
#pragma unroll
    for (int rt = 0; rt < 2; ++rt)
#pragma unroll
        for (int j = 0; j < 8; ++j)
            trow[rt][j] = tsk[rt * 16 + rbase + j];

    // y accumulators: wave owns d in [128*wave, 128*wave+128)
    const int d0 = wave * 128;
    v8f acc[2][8];
#pragma unroll
    for (int rt = 0; rt < 2; ++rt)
#pragma unroll
        for (int ct = 0; ct < 8; ++ct)
            acc[rt][ct] = (v8f)(0.0f);

    // ---- main loop over H in chunks of HC ----
    for (int kh = 0; kh < H_DIM; kh += HC) {
        // prefetch next chunk's weights toward this WGP (L2 -> WGP$)
        if (kh + HC < H_DIM) {
            const char* p1 = (const char*)(W1b + (size_t)(kh + HC + 16 * wave) * D_DIM)
                             + (size_t)lane * 64;
            __builtin_prefetch(p1, 0, 1);
            const char* p2 = (const char*)(W2b + (size_t)d0 * H_DIM + (kh + HC))
                             + (size_t)lane * 64;
            __builtin_prefetch(p2, 0, 1);
        }

        // phase 1: this wave computes h-columns [kh+16*wave, +16), K = D.
        // Software-pipelined: fragments for step k+32 load into fresh regs
        // while step k's WMMAs execute.
        v8f hacc0 = (v8f)(0.0f), hacc1 = (v8f)(0.0f);
        const __bf16* w1base = W1b + (size_t)(kh + 16 * wave) * D_DIM;
        v16bf bfc = load_b_frag(w1base, D_DIM, lane);
        v16bf a0c = load_a_frag(xb, LDX, lane);
        v16bf a1c = load_a_frag(xb + 16 * LDX, LDX, lane);
#pragma unroll 4
        for (int k = 32; k < D_DIM; k += 32) {
            v16bf bfn = load_b_frag(w1base + k, D_DIM, lane);
            v16bf a0n = load_a_frag(xb + k, LDX, lane);
            v16bf a1n = load_a_frag(xb + 16 * LDX + k, LDX, lane);
            hacc0 = wmma_bf16(a0c, bfc, hacc0);
            hacc1 = wmma_bf16(a1c, bfc, hacc1);
            bfc = bfn; a0c = a0n; a1c = a1n;
        }
        hacc0 = wmma_bf16(a0c, bfc, hacc0);
        hacc1 = wmma_bf16(a1c, bfc, hacc1);

        // bias + relu, store bf16 h chunk to LDS
        {
            const int hidx = kh + 16 * wave + col;   // global h index
            const int hloc = 16 * wave + col;        // h index within chunk
#pragma unroll
            for (int j = 0; j < 8; ++j) {
                int r0 = rbase + j;
                float v0 = hacc0[j] + b1[(size_t)trow[0][j] * H_DIM + hidx];
                hb[r0 * LDH + hloc] = (__bf16)fmaxf(v0, 0.0f);
                int r1 = 16 + rbase + j;
                float v1 = hacc1[j] + b1[(size_t)trow[1][j] * H_DIM + hidx];
                hb[r1 * LDH + hloc] = (__bf16)fmaxf(v1, 0.0f);
            }
        }
        __syncthreads();

        // phase 2: y[32 x 128(wave slice)] += h_chunk @ W2^T chunk, K = HC.
        // B fragment prefetched one column-tile ahead; A double-buffered
        // across k-steps. Fully unrolled (compile-time conditions).
        {
            v16bf a0c = load_a_frag(hb, LDH, lane);
            v16bf a1c = load_a_frag(hb + 16 * LDH, LDH, lane);
            v16bf bfc = load_b_frag(W2b + (size_t)d0 * H_DIM + kh, H_DIM, lane);
#pragma unroll
            for (int ks = 0; ks < HC / 32; ++ks) {
                const int k = ks * 32;
                v16bf a0n = a0c, a1n = a1c;
                if (ks < HC / 32 - 1) {
                    a0n = load_a_frag(hb + k + 32, LDH, lane);
                    a1n = load_a_frag(hb + 16 * LDH + k + 32, LDH, lane);
                }
#pragma unroll
                for (int ct = 0; ct < 8; ++ct) {
                    v16bf bfn = bfc;
                    if (ct < 7)
                        bfn = load_b_frag(W2b + (size_t)(d0 + (ct + 1) * 16) * H_DIM + kh + k,
                                          H_DIM, lane);
                    else if (ks < HC / 32 - 1)
                        bfn = load_b_frag(W2b + (size_t)d0 * H_DIM + kh + k + 32,
                                          H_DIM, lane);
                    acc[0][ct] = wmma_bf16(a0c, bfc, acc[0][ct]);
                    acc[1][ct] = wmma_bf16(a1c, bfc, acc[1][ct]);
                    bfc = bfn;
                }
                a0c = a0n; a1c = a1n;
            }
        }
        __syncthreads();   // hb consumed; safe to overwrite next iteration
    }

    // ---- epilogue: y = acc + x (fp32 residual) + b2[task]; spill to LDS ----
#pragma unroll
    for (int rt = 0; rt < 2; ++rt) {
#pragma unroll
        for (int ct = 0; ct < 8; ++ct) {
#pragma unroll
            for (int j = 0; j < 8; ++j) {
                int row = rt * 16 + rbase + j;
                int d   = d0 + ct * 16 + col;
                float v = acc[rt][ct][j];
                v += x[(size_t)(n0 + row) * D_DIM + d];
                v += b2[(size_t)trow[rt][j] * D_DIM + d];
                ytile[row * LDY + d] = v;   // aliases xb/hb (both dead now)
            }
        }
    }
    __syncthreads();

    // ---- LayerNorm over D per row: 8 threads per row ----
    {
        const int row = tid >> 3;          // 0..31
        const int g   = tid & 7;           // 0..7, each covers 128 d's
        const float* yr = ytile + row * LDY + g * 128;
        float s = 0.0f, ss = 0.0f;
#pragma unroll 8
        for (int i = 0; i < 128; i += 4) {
            float4 v = *(const float4*)(yr + i);
            s  += v.x + v.y + v.z + v.w;
            ss += v.x * v.x + v.y * v.y + v.z * v.z + v.w * v.w;
        }
        red[tid * 2]     = s;
        red[tid * 2 + 1] = ss;
        __syncthreads();
        if (g == 0) {
            float St = 0.0f, SSt = 0.0f;
#pragma unroll
            for (int q = 0; q < 8; ++q) {
                St  += red[(row * 8 + q) * 2];
                SSt += red[(row * 8 + q) * 2 + 1];
            }
            float mu  = St * (1.0f / D_DIM);
            float var = SSt * (1.0f / D_DIM) - mu * mu;
            musig[row * 2]     = mu;
            musig[row * 2 + 1] = rsqrtf(var + LN_EPS);
        }
        __syncthreads();
        const float mu = musig[row * 2];
        const float rs = musig[row * 2 + 1];
        const int   t  = tsk[row];
        float* orow       = out  + (size_t)(n0 + row) * D_DIM + g * 128;
        const float* wrow = ln_w + g * 128;
        const float* brow = ln_b + (size_t)t * D_DIM + g * 128;
#pragma unroll 8
        for (int i = 0; i < 128; i += 4) {
            float4 v = *(const float4*)(yr + i);
            float4 w = *(const float4*)(wrow + i);
            float4 b = *(const float4*)(brow + i);
            float4 o;
            o.x = (v.x - mu) * rs * w.x + b.x;
            o.y = (v.y - mu) * rs * w.y + b.y;
            o.z = (v.z - mu) * rs * w.z + b.z;
            o.w = (v.w - mu) * rs * w.w + b.w;
            *(float4*)(orow + i) = o;
        }
    }
}

// ---------------------------------------------------------------------------
extern "C" void kernel_launch(void* const* d_in, const int* in_sizes, int n_in,
                              void* d_out, int out_size, void* d_ws, size_t ws_size,
                              hipStream_t stream) {
    (void)in_sizes; (void)n_in; (void)out_size; (void)ws_size;
    // setup_inputs order: task_id, x, W1, b1, W2, b2, ln_w, ln_b
    const int*   task_id = (const int*)  d_in[0];
    const float* x       = (const float*)d_in[1];
    const float* W1      = (const float*)d_in[2];
    const float* b1      = (const float*)d_in[3];
    const float* W2      = (const float*)d_in[4];
    const float* b2      = (const float*)d_in[5];
    const float* ln_w    = (const float*)d_in[6];
    const float* ln_b    = (const float*)d_in[7];
    float*       out     = (float*)d_out;

    // scratch: bf16 copies of W1 [H,D] and W2 [D,H] (16.8 MB total)
    __bf16* w1b = (__bf16*)d_ws;
    __bf16* w2b = w1b + (size_t)H_DIM * D_DIM;

    {
        int n4 = (H_DIM * D_DIM) / 4;
        int blk = 256, grd = (n4 + blk - 1) / blk;
        cvt_bf16_kernel<<<grd, blk, 0, stream>>>(W1, w1b, n4);
        cvt_bf16_kernel<<<grd, blk, 0, stream>>>(W2, w2b, n4);
    }

    const int ntok = S_DIM * B_DIM;                 // 32768
    ffn_fused_kernel<<<ntok / TN, 256, SMEM_BYTES, stream>>>(
        task_id, x, b1, b2, ln_w, ln_b, w1b, w2b, out);
}
